// FullAttention_1924145349251
// MI455X (gfx1250) — matrix operand order repaired and checked by reading, not compiled, not run on hardware
//
#include <hip/hip_runtime.h>

// ---------------------------------------------------------------------------
// Causal full attention for B=4, L=2048, H=16, E=D=64 (fp32 in, fp32 out),
// returning (V, A). Tuned for MI455X (gfx1250, wave32, WMMA, 320KB LDS/WGP):
//  - f16 WMMA (v_wmma_f32_16x16x32_f16) with fp32 accumulation for both GEMMs
//  - entire 16x2048 fp32 score strip resident in LDS (128KB); softmax in fp32
//  - A written to HBM exactly once with NON-TEMPORAL stores (1.07GB stream is
//    the roofline floor; NT keeps the K/V working sets resident in the 192MB L2)
//  - causal upper triangle never computed, only zero-stored
//  - wave id read-first-laned into an SGPR so no WMMA sits under exec-masked
//    control flow (ISA: EXEC must be all-1s for WMMA)
// ---------------------------------------------------------------------------

typedef __attribute__((ext_vector_type(16))) _Float16 v16h;
typedef __attribute__((ext_vector_type(8)))  float    v8f;
typedef __attribute__((ext_vector_type(4)))  float    f32x4;

namespace {
constexpr int Bc = 4;
constexpr int Lc = 2048;
constexpr int Hc = 16;
constexpr int Ec = 64;
constexpr int Dc = 64;

constexpr int ROWS   = 16;          // query rows per workgroup
constexpr int PSTR   = Lc + 4;      // LDS row stride (floats), pad to dodge bank conflicts
constexpr int QSTR   = Ec + 8;      // LDS Q row stride (halves)
constexpr int NTHREADS = 128;       // 4 wave32

constexpr size_t P_BYTES   = (size_t)ROWS * PSTR * sizeof(float);     // 131328
constexpr size_t Q_BYTES   = (size_t)ROWS * QSTR * sizeof(_Float16);  // 2304
constexpr size_t RED_BYTES = (size_t)ROWS * 8 * sizeof(float);        // 512
constexpr size_t MAX_BYTES = (size_t)ROWS * sizeof(float);            // 64
constexpr size_t INV_BYTES = (size_t)ROWS * sizeof(float);            // 64
constexpr size_t SMEM_BYTES = P_BYTES + Q_BYTES + RED_BYTES + MAX_BYTES + INV_BYTES;
} // namespace

__global__ __launch_bounds__(NTHREADS)
void causal_attn_wmma(const float* __restrict__ q,
                      const float* __restrict__ kmat,
                      const float* __restrict__ vmat,
                      float* __restrict__ outV,
                      float* __restrict__ outA) {
    extern __shared__ char smem_raw[];
    float*    P      = reinterpret_cast<float*>(smem_raw);                       // [16][PSTR] scores -> exp(s)
    _Float16* Qh     = reinterpret_cast<_Float16*>(smem_raw + P_BYTES);          // [16][QSTR]
    float*    red    = reinterpret_cast<float*>(smem_raw + P_BYTES + Q_BYTES);   // [16][8]
    float*    rowmax = red + ROWS * 8;                                           // [16]
    float*    invsum = rowmax + ROWS;                                            // [16]

    const int tid  = threadIdx.x;
    const int lane = tid & 31;
    // Wave index as an SGPR: branches on it become scalar s_cbranch, so the
    // WMMA ops never execute under a lane-masked EXEC.
    const int wave = __builtin_amdgcn_readfirstlane(tid >> 5);
    const bool lhalf = (lane < 16);      // lanes 0-15 vs 16-31
    const int  nidx  = lane & 15;        // N (or M for A-frag) index within half

    const int mblk = blockIdx.x % (Lc / ROWS);
    const int h    = (blockIdx.x / (Lc / ROWS)) % Hc;
    const int b    = blockIdx.x / ((Lc / ROWS) * Hc);
    const int l0   = mblk * ROWS;

    const int nvalid16 = mblk + 1;             // 16-col tiles with any unmasked entry
    const int npv16    = (nvalid16 + 1) & ~1;  // round up so coverage is a multiple of 32
    const int ncols    = npv16 * 16;           // LDS columns that hold defined data
    const float scale  = 0.125f;               // 1/sqrt(64)
    const float NEG_INF = -__builtin_inff();

    // ---- Phase 0: stage Q tile (16x64) into LDS as f16; pad-fill the (at
    //      most one) extra 16-col tile so PV runs in clean 32-wide chunks -----
    for (int i = tid; i < ROWS * Ec; i += NTHREADS) {
        const int m = i >> 6, e = i & 63;
        Qh[m * QSTR + e] =
            (_Float16)q[(((size_t)b * Lc + (l0 + m)) * Hc + h) * Ec + e];
    }
    if (npv16 != nvalid16) {                   // block-uniform condition
        const int s0 = nvalid16 * 16;
        for (int i = tid; i < ROWS * 16; i += NTHREADS)
            P[(i >> 4) * PSTR + s0 + (i & 15)] = NEG_INF;
    }
    __syncthreads();

    // ---- Build per-wave A-fragments of Q (16x32 halves each; E=64 -> two) ---
    v16h aQ0, aQ1;
#pragma unroll
    for (int i = 0; i < 8; ++i) {
        const int kk = (lhalf ? 0 : 16) + 2 * i;
        aQ0[2 * i]     = Qh[nidx * QSTR + kk];
        aQ0[2 * i + 1] = Qh[nidx * QSTR + kk + 1];
        aQ1[2 * i]     = Qh[nidx * QSTR + 32 + kk];
        aQ1[2 * i + 1] = Qh[nidx * QSTR + 32 + kk + 1];
    }

    // ---- Phase 1: scores = scale * Q K^T (causal masked) into LDS -----------
    const int mrow = lhalf ? 0 : 8;   // C-frag: VGPR r -> M = r + mrow
    for (int jt = wave; jt < nvalid16; jt += 4) {
        const int j = jt * 16 + nidx;   // key row for this lane's B column
        const float* kp = kmat + (((size_t)b * Lc + j) * Hc + h) * Ec;
        v16h b0, b1;
#pragma unroll
        for (int i = 0; i < 8; ++i) {
            const int kk = (lhalf ? 0 : 16) + 2 * i;   // B-frag K enumeration
            const float2 f0 = *reinterpret_cast<const float2*>(kp + kk);
            const float2 f1 = *reinterpret_cast<const float2*>(kp + 32 + kk);
            b0[2 * i] = (_Float16)f0.x;  b0[2 * i + 1] = (_Float16)f0.y;
            b1[2 * i] = (_Float16)f1.x;  b1[2 * i + 1] = (_Float16)f1.y;
        }
        v8f c = {};
        c = __builtin_amdgcn_wmma_f32_16x16x32_f16(false, aQ0, false, b0,
                                                   (short)0, c, false, false);
        c = __builtin_amdgcn_wmma_f32_16x16x32_f16(false, aQ1, false, b1,
                                                   (short)0, c, false, false);
#pragma unroll
        for (int r = 0; r < 8; ++r) {
            const int m = r + mrow;
            const int s = jt * 16 + nidx;
            P[m * PSTR + s] = (s <= l0 + m) ? (scale * c[r]) : NEG_INF;
        }
    }
    __syncthreads();

    // ---- Phase 2: fp32 softmax over LDS strip (8 threads per row) -----------
    {
        const int row = tid >> 3, sub = tid & 7;
        float mx = NEG_INF;
        for (int s = sub; s < ncols; s += 8) mx = fmaxf(mx, P[row * PSTR + s]);
        red[row * 8 + sub] = mx;
        __syncthreads();
        if (sub == 0) {
            float m2 = red[row * 8];
#pragma unroll
            for (int i = 1; i < 8; ++i) m2 = fmaxf(m2, red[row * 8 + i]);
            rowmax[row] = m2;
        }
        __syncthreads();
        const float rm = rowmax[row];
        float ss = 0.0f;
        for (int s = sub; s < ncols; s += 8) {
            const float e = __expf(P[row * PSTR + s] - rm);
            P[row * PSTR + s] = e;      // strip now holds unnormalized exp()
            ss += e;
        }
        red[row * 8 + sub] = ss;
        __syncthreads();
        if (sub == 0) {
            float t = 0.0f;
#pragma unroll
            for (int i = 0; i < 8; ++i) t += red[row * 8 + i];
            invsum[row] = 1.0f / t;
        }
        __syncthreads();
    }

    // ---- Phase 3: store A (normalized). Non-temporal: this 1GB stream is
    //      write-once and must not evict the L2-resident K/V working sets. ----
    {
        float* arow = outA + (((size_t)b * Hc + h) * Lc + l0) * (size_t)Lc;
        for (int idx = tid; idx < ROWS * (Lc / 4); idx += NTHREADS) {
            const int row = idx / (Lc / 4);
            const int s   = (idx % (Lc / 4)) * 4;
            f32x4 o;
            if (s < ncols) {
                const f32x4 pv = *reinterpret_cast<const f32x4*>(&P[row * PSTR + s]);
                o = pv * invsum[row];
            } else {
                o = (f32x4)0.0f;                          // masked region
            }
            __builtin_nontemporal_store(
                o, reinterpret_cast<f32x4*>(arow + (size_t)row * Lc + s));
        }
    }

    // ---- Phase 4: V_out = (P_exp · V) * invsum ; wave w owns D-tile w -------
    {
        const int dcol = wave * 16 + nidx;
        v8f acc = {};
        for (int sc = 0; sc < ncols; sc += 32) {
            v16h aP, bV;
#pragma unroll
            for (int i = 0; i < 8; ++i) {
                const int ka = (lhalf ? 0 : 16) + 2 * i;
                aP[2 * i]     = (_Float16)P[nidx * PSTR + sc + ka];
                aP[2 * i + 1] = (_Float16)P[nidx * PSTR + sc + ka + 1];
                const int s0 = sc + (lhalf ? 0 : 16) + 2 * i;
                bV[2 * i]     = (_Float16)vmat[(((size_t)b * Lc + s0) * Hc + h) * Dc + dcol];
                bV[2 * i + 1] = (_Float16)vmat[(((size_t)b * Lc + s0 + 1) * Hc + h) * Dc + dcol];
            }
            acc = __builtin_amdgcn_wmma_f32_16x16x32_f16(false, aP, false, bV,
                                                         (short)0, acc, false, false);
        }
#pragma unroll
        for (int r = 0; r < 8; ++r) {
            const int m = r + mrow;
            outV[(((size_t)b * Lc + (l0 + m)) * Hc + h) * Dc + dcol] =
                acc[r] * invsum[m];
        }
    }
}

extern "C" void kernel_launch(void* const* d_in, const int* in_sizes, int n_in,
                              void* d_out, int out_size, void* d_ws, size_t ws_size,
                              hipStream_t stream) {
    (void)in_sizes; (void)n_in; (void)out_size; (void)d_ws; (void)ws_size;
    const float* q = (const float*)d_in[0];
    const float* k = (const float*)d_in[1];
    const float* v = (const float*)d_in[2];
    float* outV = (float*)d_out;                                  // [B,L,H,D]
    float* outA = outV + (size_t)Bc * Lc * Hc * Dc;               // [B,H,L,S]

    const dim3 grid(Bc * Hc * (Lc / ROWS));                       // 8192 blocks
    const dim3 block(NTHREADS);                                   // 4 wave32
    causal_attn_wmma<<<grid, block, SMEM_BYTES, stream>>>(q, k, v, outV, outA);
}
